// Attn_89756226552074
// MI455X (gfx1250) — matrix operand order repaired and checked by reading, not compiled, not run on hardware
//
#include <hip/hip_runtime.h>
#include <math.h>

#define B_  64
#define S_  2048
#define K_  512
#define A_  512
#define V_  512
#define Q_  512

typedef __attribute__((ext_vector_type(16))) _Float16 v16h;
typedef __attribute__((ext_vector_type(8)))  _Float16 v8h;
typedef __attribute__((ext_vector_type(4)))  _Float16 v4h;
typedef __attribute__((ext_vector_type(8)))  float    v8f;
typedef __attribute__((ext_vector_type(4)))  float    vf4;
typedef __attribute__((ext_vector_type(2)))  float    vf2;

// Hardware tanh (CDNA5 TRANS op). One independent instr required after a
// TRANS op before its result is consumed (ISA 7.4) -> append v_nop since the
// compiler cannot see inside the asm to schedule the gap itself.
__device__ inline float htanh(float x) {
  float r;
  asm("v_tanh_f32 %0, %1\n\tv_nop" : "=v"(r) : "v"(x));
  return r;
}

// ---------------- Kernel 0: pack Wk (f32 -> f16), row-major [A,K] ----------
__global__ __launch_bounds__(256) void wkpack_kernel(const float* __restrict__ Wk,
                                                     _Float16* __restrict__ wkh) {
  int i = (blockIdx.x * 256 + threadIdx.x) * 4;   // A_*K_ / 4 threads
  vf4 f = *(const vf4*)(Wk + i);
  v4h h;
  h[0] = (_Float16)f[0]; h[1] = (_Float16)f[1];
  h[2] = (_Float16)f[2]; h[3] = (_Float16)f[3];
  *(v4h*)(wkh + i) = h;
}

// ---------------- Kernel 1: qp[b,a] = q[b,:]·Wq[a,:] + bq[a] ----------------
__global__ __launch_bounds__(256) void qproj_kernel(const float* __restrict__ q,
                                                    const float* __restrict__ Wq,
                                                    const float* __restrict__ bq,
                                                    float* __restrict__ qp) {
  int idx = blockIdx.x * blockDim.x + threadIdx.x;   // B_*A_ = 32768 threads
  int b = idx >> 9;          // / A_
  int a = idx & (A_ - 1);
  const float* qr = q  + (size_t)b * Q_;
  const float* wr = Wq + (size_t)a * Q_;
  float acc = 0.f;
  #pragma unroll 4
  for (int kk = 0; kk < Q_; kk += 4) {
    vf4 qa = *(const vf4*)(qr + kk);
    vf4 wa = *(const vf4*)(wr + kk);
    acc += qa[0]*wa[0] + qa[1]*wa[1] + qa[2]*wa[2] + qa[3]*wa[3];
  }
  qp[idx] = acc + bq[a];
}

// ---- Kernel 2: fused  energy[b,s] = Wa·tanh(k[b,s]·Wkᵀ + bk + qp[b]) + ba --
// Block: 256 threads (8 waves). Covers 64 rows (2 pairs of M-tiles, same b)
// times all 512 A-columns (wave w owns cols [64w, 64w+64) = 4 N-tiles).
// Each B fragment (f16, prepacked) feeds 2 WMMAs (both M-tiles of the pair).
__global__ __launch_bounds__(256) void energy_kernel(
    const float* __restrict__ k,
    const _Float16* __restrict__ wkh,
    const float* __restrict__ bk,
    const float* __restrict__ Wa,
    const float* __restrict__ ba,
    const float* __restrict__ qp,
    float* __restrict__ energy) {
  __shared__ float e_lds[64];
  const int tid  = threadIdx.x;
  const int wave = tid >> 5;
  const int lane = tid & 31;
  const int half = lane >> 4;       // 0: lanes 0-15, 1: lanes 16-31
  const int lr   = lane & 15;

  if (tid < 64) e_lds[tid] = 0.f;
  __syncthreads();

  const int row0 = blockIdx.x * 64;     // 64 flattened (b,s) rows, same b
  const int b    = row0 >> 11;          // / S_

  // Per-lane per-N-tile constants: qb = qp[b,a] + bk[a], wav = Wa[a]
  const int colbase = wave * 64;
  float qb[4], wav[4];
  #pragma unroll
  for (int nt = 0; nt < 4; ++nt) {
    int a = colbase + nt * 16 + lr;
    qb[nt]  = qp[(size_t)b * A_ + a] + bk[a];
    wav[nt] = Wa[a];
  }

  for (int mp = 0; mp < 2; ++mp) {      // pair of M-tiles (32 rows)
    const int rowg = row0 + mp * 32;
    v8f acc[2][4];
    #pragma unroll
    for (int m = 0; m < 2; ++m)
      #pragma unroll
      for (int nt = 0; nt < 4; ++nt)
        #pragma unroll
        for (int r = 0; r < 8; ++r) acc[m][nt][r] = 0.f;

    #pragma unroll 1
    for (int ks = 0; ks < 16; ++ks) {
      const int kb = ks * 32;
      // ---- Two A fragments (16x32 f16): lane(l<16) row=l holds K c0..c0+7
      //      and c0+16..c0+23 with c0 = 8*half (ISA 16-bit A layout).
      v16h af[2];
      #pragma unroll
      for (int m = 0; m < 2; ++m) {
        const float* ap = k + (size_t)(rowg + m * 16 + lr) * K_ + kb + half * 8;
        vf4 a0 = *(const vf4*)(ap);
        vf4 a1 = *(const vf4*)(ap + 4);
        vf4 a2 = *(const vf4*)(ap + 16);
        vf4 a3 = *(const vf4*)(ap + 20);
        #pragma unroll
        for (int i = 0; i < 4; ++i) {
          af[m][i]      = (_Float16)a0[i];
          af[m][4 + i]  = (_Float16)a1[i];
          af[m][8 + i]  = (_Float16)a2[i];
          af[m][12 + i] = (_Float16)a3[i];
        }
      }
      #pragma unroll
      for (int nt = 0; nt < 4; ++nt) {
        // ---- B fragment (32x16 f16): col a = colbase+nt*16+lr,
        //      lane half h holds K kb+8h..kb+8h+7 and kb+16+8h..kb+16+8h+7 (same as A).
        //      Prepacked f16 -> two b128 loads, no conversion.
        const _Float16* bp = wkh + (size_t)(colbase + nt * 16 + lr) * K_ + kb + half * 8;
        v8h b0 = *(const v8h*)(bp);
        v8h b1 = *(const v8h*)(bp + 16);
        v16h bf = __builtin_shufflevector(b0, b1, 0, 1, 2, 3, 4, 5, 6, 7,
                                          8, 9, 10, 11, 12, 13, 14, 15);
        acc[0][nt] = __builtin_amdgcn_wmma_f32_16x16x32_f16(
            false, af[0], false, bf, (short)0, acc[0][nt], false, false);
        acc[1][nt] = __builtin_amdgcn_wmma_f32_16x16x32_f16(
            false, af[1], false, bf, (short)0, acc[1][nt], false, false);
      }
    }

    // Fused epilogue: hidden = tanh(kp + bk + qp); partial energy = Wa*hidden.
    // C layout: VGPR r on lane l -> row r + 8*half, col = l&15.
    #pragma unroll
    for (int m = 0; m < 2; ++m) {
      float part[8];
      #pragma unroll
      for (int r = 0; r < 8; ++r) part[r] = 0.f;
      #pragma unroll
      for (int nt = 0; nt < 4; ++nt)
        #pragma unroll
        for (int r = 0; r < 8; ++r)
          part[r] += wav[nt] * htanh(acc[m][nt][r] + qb[nt]);

      // Reduce over the 16 columns held by each 16-lane half.
      #pragma unroll
      for (int r = 0; r < 8; ++r) {
        float s = part[r];
        s += __shfl_xor(s, 1, 32);
        s += __shfl_xor(s, 2, 32);
        s += __shfl_xor(s, 4, 32);
        s += __shfl_xor(s, 8, 32);
        part[r] = s;
      }
      if (lr == 0) {
        #pragma unroll
        for (int r = 0; r < 8; ++r)
          atomicAdd(&e_lds[(mp * 2 + m) * 16 + half * 8 + r], part[r]);  // ds_add_f32
      }
    }
  }
  __syncthreads();
  if (tid < 64) energy[row0 + tid] = e_lds[tid] + ba[0];
}

// ---------------- Kernel 3: softmax over S per batch ----------------
__global__ __launch_bounds__(256) void softmax_kernel(
    const float* __restrict__ energy, float* __restrict__ score) {
  __shared__ float red[8];
  const int b   = blockIdx.x;
  const int tid = threadIdx.x;
  const float* e = energy + (size_t)b * S_;
  float vals[8];
  float lmax = -3.4e38f;
  #pragma unroll
  for (int i = 0; i < 8; ++i) {
    vals[i] = e[tid + i * 256];
    lmax = fmaxf(lmax, vals[i]);
  }
  for (int m = 1; m < 32; m <<= 1) lmax = fmaxf(lmax, __shfl_xor(lmax, m, 32));
  if ((tid & 31) == 0) red[tid >> 5] = lmax;
  __syncthreads();
  float gmax = red[0];
  #pragma unroll
  for (int w = 1; w < 8; ++w) gmax = fmaxf(gmax, red[w]);
  __syncthreads();
  float lsum = 0.f;
  #pragma unroll
  for (int i = 0; i < 8; ++i) {
    vals[i] = __expf(vals[i] - gmax);
    lsum += vals[i];
  }
  for (int m = 1; m < 32; m <<= 1) lsum += __shfl_xor(lsum, m, 32);
  if ((tid & 31) == 0) red[tid >> 5] = lsum;
  __syncthreads();
  float gsum = 0.f;
  #pragma unroll
  for (int w = 0; w < 8; ++w) gsum += red[w];
  const float inv = 1.0f / gsum;
  #pragma unroll
  for (int i = 0; i < 8; ++i)
    score[(size_t)b * S_ + tid + i * 256] = vals[i] * inv;
}

// ---------------- Kernel 4: attn[b,v] = sum_s score[b,s] * v[b,s,v] --------
// One block per batch; 256 threads x float2 cover all 512 V columns with
// coalesced b64 loads (v is half of total HBM traffic -> widest useful loads).
__global__ __launch_bounds__(256) void attnv_kernel(
    const float* __restrict__ score, const float* __restrict__ v,
    float* __restrict__ attn) {
  __shared__ float sc[S_];
  const int b  = blockIdx.x;
  const int vc = threadIdx.x * 2;
  #pragma unroll
  for (int i = 0; i < 8; ++i)
    sc[threadIdx.x + i * 256] = score[(size_t)b * S_ + threadIdx.x + i * 256];
  __syncthreads();
  const float* vb = v + (size_t)b * S_ * V_ + vc;
  float acc0 = 0.f, acc1 = 0.f;
  #pragma unroll 4
  for (int s = 0; s < S_; ++s) {
    vf2 vv = *(const vf2*)(vb + (size_t)s * V_);
    float sw = sc[s];
    acc0 += sw * vv[0];
    acc1 += sw * vv[1];
  }
  vf2 out; out[0] = acc0; out[1] = acc1;
  *(vf2*)(attn + (size_t)b * V_ + vc) = out;
}

extern "C" void kernel_launch(void* const* d_in, const int* in_sizes, int n_in,
                              void* d_out, int out_size, void* d_ws, size_t ws_size,
                              hipStream_t stream) {
  const float* k  = (const float*)d_in[0];
  const float* v  = (const float*)d_in[1];
  const float* q  = (const float*)d_in[2];
  const float* Wk = (const float*)d_in[3];
  const float* bk = (const float*)d_in[4];
  const float* Wq = (const float*)d_in[5];
  const float* bq = (const float*)d_in[6];
  const float* Wa = (const float*)d_in[7];
  const float* ba = (const float*)d_in[8];

  float* score = (float*)d_out;                 // [B,S]
  float* attn  = score + (size_t)B_ * S_;       // [B,V] (reference: [B,1,V])

  float*    qp     = (float*)d_ws;                      // [B,A]  = 128 KB
  float*    energy = qp + (size_t)B_ * A_;              // [B,S]  = 512 KB
  _Float16* wkh    = (_Float16*)(energy + (size_t)B_ * S_);  // [A,K] f16 = 512 KB

  wkpack_kernel <<<(A_ * K_) / 1024, 256, 0, stream>>>(Wk, wkh);
  qproj_kernel  <<<(B_ * A_) / 256,  256, 0, stream>>>(q, Wq, bq, qp);
  energy_kernel <<<(B_ * S_) / 64,   256, 0, stream>>>(k, wkh, bk, Wa, ba, qp, energy);
  softmax_kernel<<<B_,               256, 0, stream>>>(energy, score);
  attnv_kernel  <<<B_,               256, 0, stream>>>(score, v, attn);
}
